// BoundaryPredictor3_62199716380912
// MI455X (gfx1250) — matrix operand order repaired and checked by reading, not compiled, not run on hardware
//
#include <hip/hip_runtime.h>
#include <hip/hip_bf16.h>
#include <math.h>
#include <cstdint>

// ---------------------------------------------------------------------------
// BoundaryPredictor forward for MI455X (gfx1250, wave32).
// Heavy GEMMs (16384x512x512) run on v_wmma_f32_16x16x32_bf16 with the
// 64x512 bf16 weight strip DMA'd into LDS per block via the Tensor Data
// Mover (tensor_load_to_lds + s_wait_tensorcnt), padded to avoid LDS bank
// conflicts on the B-fragment ds_load_b128s.
// ---------------------------------------------------------------------------

typedef __bf16 bf16_t;
typedef __attribute__((ext_vector_type(16))) __bf16 v16bf;
typedef __attribute__((ext_vector_type(8)))  __bf16 v8bf;
typedef __attribute__((ext_vector_type(8)))  float  v8f;
typedef __attribute__((ext_vector_type(4)))  unsigned int v4u;
typedef __attribute__((ext_vector_type(8)))  int v8i;
typedef __attribute__((ext_vector_type(4)))  int v4i;

#define BB 8
#define LL 2048
#define GK 512          // GEMM K (= D)
#define GN 512          // GEMM N (= D)
#define BROW 520        // LDS row stride in bf16 elems (512 + 4-dword TDM pad)

#if defined(__has_include)
#if __has_include(<hip/amd_detail/amd_gfx1250_TDM.h>)
#define TDM_6ARG 1
#endif
#endif

// ---- 1) per-row L2 norm, convert hidden & normalized hidden to bf16 --------
__global__ __launch_bounds__(256)
void rownorm_cvt_kernel(const float* __restrict__ H, bf16_t* __restrict__ Hb,
                        bf16_t* __restrict__ NHb, float* __restrict__ rnorm, int D) {
  int row = blockIdx.x;
  const float* h = H + (size_t)row * D;
  __shared__ float red[256];
  float ss = 0.f;
  for (int d = threadIdx.x; d < D; d += 256) { float v = h[d]; ss += v * v; }
  red[threadIdx.x] = ss; __syncthreads();
  for (int s = 128; s > 0; s >>= 1) {
    if (threadIdx.x < s) red[threadIdx.x] += red[threadIdx.x + s];
    __syncthreads();
  }
  float inv = 1.f / fmaxf(sqrtf(red[0]), 1e-12f);
  if (threadIdx.x == 0) rnorm[row] = inv;
  for (int d = threadIdx.x; d < D; d += 256) {
    float v = h[d];
    Hb[(size_t)row * D + d]  = (bf16_t)v;
    NHb[(size_t)row * D + d] = (bf16_t)(v * inv);
  }
}

// ---- 2) Mt[j,i] = (Wqb^T Wkb)[i,j] = sum_n Wqb[n,i]*Wkb[n,j]  (bf16) -------
__global__ __launch_bounds__(256)
void mt_prep_kernel(const float* __restrict__ Wqb, const float* __restrict__ Wkb,
                    bf16_t* __restrict__ Mt, int D) {
  int idx = blockIdx.x * 256 + threadIdx.x;
  if (idx >= D * D) return;
  int j = idx / D, i = idx - j * D;
  float s = 0.f;
  for (int n = 0; n < D; ++n) s += Wqb[(size_t)n * D + i] * Wkb[(size_t)n * D + j];
  Mt[idx] = (bf16_t)s;
}

__global__ __launch_bounds__(256)
void cvt_bf16_kernel(const float* __restrict__ X, bf16_t* __restrict__ Y, int n) {
  for (int i = blockIdx.x * 256 + threadIdx.x; i < n; i += gridDim.x * 256)
    Y[i] = (bf16_t)X[i];
}

// ---- 3) C[M,512] = A[M,512](bf16) * W[512,512](bf16)^T via WMMA ------------
// block = 256 threads = 8 waves; each wave one 16x64 tile over the block's
// 64-wide N strip. The strip's weights are TDM-DMA'd into LDS once per block.
__global__ __launch_bounds__(256)
void gemm_bf16_wmma_kernel(const bf16_t* __restrict__ A, const bf16_t* __restrict__ W,
                           float* __restrict__ C, int M) {
  __shared__ bf16_t Bs[64 * BROW];       // 65 KB padded weight strip [n_local][k]
  const int lane = threadIdx.x & 31;
  const int wave = threadIdx.x >> 5;
  const int half = lane >> 4;            // 0: lanes 0-15, 1: lanes 16-31
  const int r    = lane & 15;
  const int m0 = (blockIdx.y * 8 + wave) * 16;
  const int n0 = blockIdx.x * 64;

  // ---- TDM: DMA 64 x 512 bf16 strip (row-major, K contiguous) into LDS ----
  if (wave == 0) {
    const unsigned long long ga =
        (unsigned long long)(uintptr_t)(W + (size_t)n0 * GK);
    const unsigned int lds = (unsigned int)(uintptr_t)(&Bs[0]); // addr[31:0]=LDS off
    v4u g0;
    g0[0] = 1u;                                    // count=1 (valid), no gather
    g0[1] = lds;                                   // lds_addr
    g0[2] = (unsigned int)(ga & 0xffffffffu);      // global_addr[31:0]
    g0[3] = (unsigned int)((ga >> 32) & 0x01ffffffu) | (2u << 30); // [56:32]|type=2
    v8i g1;
    g1[0] = (int)((1u << 16)      // data_size = 1 (2 bytes)
                | (1u << 20)      // pad_enable
                | (7u << 22)      // pad_interval: every 256 dwords (one row)
                | (3u << 25));    // pad_amount: 4 dwords -> BROW=520 elems
    g1[1] = (int)((unsigned)GK << 16);   // tensor_dim0[15:0] in bits 63:48
    g1[2] = (int)((unsigned)GN << 16);   // dim0 hi=0 | tensor_dim1[15:0]
    g1[3] = (int)((unsigned)GK << 16);   // dim1 hi=0 | tile_dim0 = 512
    g1[4] = 64;                          // tile_dim1 = 64, tile_dim2 = 0
    g1[5] = GK;                          // tensor_dim0_stride[31:0]
    g1[6] = (int)((unsigned)GK << 16);   // stride0 hi=0 | dim1_stride[15:0]
    g1[7] = 0;
    v4i g2 = {};
    v4i g3 = {};
#ifdef TDM_6ARG
    v8i g4 = {};
    __builtin_amdgcn_tensor_load_to_lds(g0, g1, g2, g3, g4, 0);
#else
    __builtin_amdgcn_tensor_load_to_lds(g0, g1, g2, g3, 0);
#endif
    __builtin_amdgcn_s_wait_tensorcnt(0);
  }
  __syncthreads();
  if (m0 >= M) return;

  const bf16_t* Arow = A + (size_t)(m0 + r) * GK;
  v8f acc[4] = {};
  for (int k0 = 0; k0 < GK; k0 += 32) {
    // A fragment: lane half 0 holds K {0..7,16..23}, half 1 holds {8..15,24..31}
    const int ka = k0 + half * 8;
    v16bf afrag;
    {
      v8bf lo = *(const v8bf*)(Arow + ka);
      v8bf hi = *(const v8bf*)(Arow + ka + 16);
#pragma unroll
      for (int i = 0; i < 8; ++i) { afrag[i] = lo[i]; afrag[i + 8] = hi[i]; }
    }
    if (k0 + 32 < GK) __builtin_prefetch(Arow + k0 + 32, 0, 1); // global_prefetch_b8
#pragma unroll
    for (int t = 0; t < 4; ++t) {
      // B fragment from LDS: lane holds column n = t*16 + r, K {0..7,16..23} (half 0) / {8..15,24..31} (half 1)
      const bf16_t* Bl = Bs + (size_t)(t * 16 + r) * BROW + k0 + half * 8;
      v8bf lo = *(const v8bf*)(Bl);
      v8bf hi = *(const v8bf*)(Bl + 16);
      v16bf bfrag;
#pragma unroll
      for (int i = 0; i < 8; ++i) { bfrag[i] = lo[i]; bfrag[i + 8] = hi[i]; }
      acc[t] = __builtin_amdgcn_wmma_f32_16x16x32_bf16(
          /*neg_a=*/false, afrag, /*neg_b=*/false, bfrag,
          /*c_mod=*/(short)0, acc[t], /*reuse_a=*/false, /*reuse_b=*/false);
    }
  }
  // C layout: VGPR j -> row m0 + j + half*8, col n0 + t*16 + (lane&15)
#pragma unroll
  for (int t = 0; t < 4; ++t) {
    const int n = n0 + t * 16 + r;
#pragma unroll
    for (int j = 0; j < 8; ++j) {
      const int m = m0 + j + half * 8;
      C[(size_t)m * GN + n] = acc[t][j];
    }
  }
}

// ---- 4) boundary probs -> hard bits (wave per token) -----------------------
// cos[b,l] = dot(y[b,l-1], hidden[b,l]) * rnorm[b,l] ; probs[b,0] = 1
__global__ __launch_bounds__(256)
void boundary_hard_kernel(const float* __restrict__ y, const float* __restrict__ H,
                          const float* __restrict__ rnorm,
                          const float* __restrict__ noise_u,
                          int* __restrict__ hard, int D) {
  const int token = blockIdx.x * 8 + (threadIdx.x >> 5);
  const int lane  = threadIdx.x & 31;
  if (token >= BB * LL) return;
  const int l = token & (LL - 1);
  float p;
  if (l == 0) {
    p = 1.0f;
  } else {
    const float* yv = y + (size_t)(token - 1) * D;
    const float* hv = H + (size_t)token * D;
    float acc = 0.f;
    for (int i = lane; i < D; i += 32) acc += yv[i] * hv[i];
    for (int off = 16; off > 0; off >>= 1) acc += __shfl_down(acc, off, 32);
    float cs = acc * rnorm[token];
    p = fminf(fmaxf((1.f - cs) * 0.5f, 0.f), 1.f);
  }
  if (lane == 0) {
    float pc = fminf(fmaxf(p, 1e-6f), 1.f - 1e-6f);
    float logits = logf(pc) - log1pf(-pc);
    float u = noise_u[token];
    float noise = logf(u) - log1pf(-u);
    // sigmoid((logits+noise)/T) > 0.5  <=>  logits + noise > 0  (T = 1)
    hard[token] = ((logits + noise) > 0.f) ? 1 : 0;
  }
}

// ---- 5) per-batch inclusive scan of hard bits -> segment starts ------------
__global__ __launch_bounds__(1024)
void seg_scan_kernel(const int* __restrict__ hard, int* __restrict__ seg_start,
                     int* __restrict__ nseg) {
  const int b = blockIdx.x;
  __shared__ int sh[2][LL];
  const int* hb = hard + (size_t)b * LL;
  for (int i = threadIdx.x; i < LL; i += blockDim.x) sh[0][i] = hb[i];
  __syncthreads();
  int cur = 0;
  for (int off = 1; off < LL; off <<= 1) {
    for (int i = threadIdx.x; i < LL; i += blockDim.x) {
      int v = sh[cur][i];
      if (i >= off) v += sh[cur][i - off];
      sh[cur ^ 1][i] = v;
    }
    __syncthreads();
    cur ^= 1;
  }
  const int total = sh[cur][LL - 1];
  for (int i = threadIdx.x; i < LL; i += blockDim.x) {
    if (hb[i]) seg_start[(size_t)b * (LL + 1) + (sh[cur][i] - 1)] = i;
  }
  if (threadIdx.x == 0) {
    seg_start[(size_t)b * (LL + 1) + total] = LL;
    nseg[b] = total;
  }
}

// ---- 6) segment means (bf16, zeros for empty segments) ---------------------
__global__ __launch_bounds__(128)
void seg_mean_kernel(const float* __restrict__ H, const int* __restrict__ seg_start,
                     const int* __restrict__ nseg, bf16_t* __restrict__ SMb, int D) {
  const int b = blockIdx.y, s = blockIdx.x;
  bf16_t* out = SMb + ((size_t)b * LL + s) * D;
  if (s >= nseg[b]) {
    for (int d = threadIdx.x; d < D; d += 128) out[d] = (bf16_t)0.f;
    return;
  }
  const int st = seg_start[(size_t)b * (LL + 1) + s];
  const int en = seg_start[(size_t)b * (LL + 1) + s + 1];
  const float inv = 1.f / (float)(en - st);
  for (int d = threadIdx.x; d < D; d += 128) {
    float acc = 0.f;
    for (int l = st; l < en; ++l) acc += H[((size_t)b * LL + l) * D + d];
    out[d] = (bf16_t)(acc * inv);
  }
}

// ---- 7) segment-local softmax attention pool -------------------------------
__global__ __launch_bounds__(256)
void attn_pool_kernel(const float* __restrict__ Q, const float* __restrict__ Km,
                      const float* __restrict__ Vm, const int* __restrict__ seg_start,
                      const int* __restrict__ nseg, float* __restrict__ out, int D) {
  const int b = blockIdx.y, s = blockIdx.x;
  float* o = out + ((size_t)b * LL + s) * D;
  if (s >= nseg[b]) {
    for (int d = threadIdx.x; d < D; d += 256) o[d] = 0.f;  // counts>0 mask
    return;
  }
  const int st = seg_start[(size_t)b * (LL + 1) + s];
  const int en = seg_start[(size_t)b * (LL + 1) + s + 1];
  const int len = en - st;
  __shared__ float sc[LL];
  __shared__ float red[256];
  const float* q = Q + ((size_t)b * LL + s) * D;
  const int lane = threadIdx.x & 31, wave = threadIdx.x >> 5;
  const float scale = rsqrtf((float)D);
  for (int li = wave; li < len; li += 8) {
    const float* k = Km + ((size_t)b * LL + st + li) * D;
    float acc = 0.f;
    for (int i = lane; i < D; i += 32) acc += q[i] * k[i];
    for (int off = 16; off > 0; off >>= 1) acc += __shfl_down(acc, off, 32);
    if (lane == 0) sc[li] = acc * scale;
  }
  __syncthreads();
  // softmax over the segment (masked-out positions contribute exp(-1e9)==0)
  float m = -1e30f;
  for (int i = threadIdx.x; i < len; i += 256) m = fmaxf(m, sc[i]);
  red[threadIdx.x] = m; __syncthreads();
  for (int s2 = 128; s2 > 0; s2 >>= 1) {
    if (threadIdx.x < s2) red[threadIdx.x] = fmaxf(red[threadIdx.x], red[threadIdx.x + s2]);
    __syncthreads();
  }
  m = red[0]; __syncthreads();
  float ssum = 0.f;
  for (int i = threadIdx.x; i < len; i += 256) {
    float e = __expf(sc[i] - m); sc[i] = e; ssum += e;
  }
  red[threadIdx.x] = ssum; __syncthreads();
  for (int s2 = 128; s2 > 0; s2 >>= 1) {
    if (threadIdx.x < s2) red[threadIdx.x] += red[threadIdx.x + s2];
    __syncthreads();
  }
  const float inv = 1.f / red[0];
  for (int d = threadIdx.x; d < D; d += 256) {
    float acc = 0.f;
    for (int li = 0; li < len; ++li)
      acc += sc[li] * Vm[((size_t)b * LL + st + li) * D + d];
    o[d] = acc * inv;
  }
}

// ---- 8) binomial prior loss + counters -------------------------------------
__global__ void loss_kernel(const int* __restrict__ nseg, float* __restrict__ out3) {
  if (threadIdx.x != 0 || blockIdx.x != 0) return;
  const float n = (float)LL;
  const float logp = logf(0.2f), log1mp = log1pf(-0.2f);
  float acc = 0.f, total = 0.f;
  for (int b = 0; b < BB; ++b) {
    float k = (float)nseg[b];
    total += k;
    float lp = lgammaf(n + 1.f) - lgammaf(k + 1.f) - lgammaf(n - k + 1.f)
             + k * logp + (n - k) * log1mp;
    acc += lp;
  }
  out3[0] = -(acc / (float)BB) / n;   // loss
  out3[1] = total;                    // num_boundaries
  out3[2] = (float)(BB * LL);         // total_positions
}

// ---------------------------------------------------------------------------
extern "C" void kernel_launch(void* const* d_in, const int* in_sizes, int n_in,
                              void* d_out, int out_size, void* d_ws, size_t ws_size,
                              hipStream_t stream) {
  const float* hidden  = (const float*)d_in[0];
  const float* noise_u = (const float*)d_in[1];
  const float* Wqb     = (const float*)d_in[2];
  const float* Wkb     = (const float*)d_in[3];
  const float* Wq      = (const float*)d_in[4];
  const float* Wk      = (const float*)d_in[5];
  const float* Wv      = (const float*)d_in[6];
  float* out = (float*)d_out;

  const int B = BB, L = LL;
  const int BLn = B * L;                    // 16384
  const int D = in_sizes[0] / BLn;          // 512

  char* ws = (char*)d_ws;
  size_t off = 0;
  auto take = [&](size_t bytes) -> char* {
    off = (off + 255) & ~(size_t)255;
    char* p = ws + off;
    off += bytes;
    return p;
  };
  bf16_t* hid_bf   = (bf16_t*)take((size_t)BLn * D * 2);
  bf16_t* nh_bf    = (bf16_t*)take((size_t)BLn * D * 2);
  float*  rnorm    = (float*) take((size_t)BLn * 4);
  bf16_t* Mt_bf    = (bf16_t*)take((size_t)D * D * 2);
  bf16_t* Wq_bf    = (bf16_t*)take((size_t)D * D * 2);
  bf16_t* Wk_bf    = (bf16_t*)take((size_t)D * D * 2);
  bf16_t* Wv_bf    = (bf16_t*)take((size_t)D * D * 2);
  float*  yb       = (float*) take((size_t)BLn * D * 4);
  float*  Kmat     = (float*) take((size_t)BLn * D * 4);
  float*  Vmat     = (float*) take((size_t)BLn * D * 4);
  float*  Qmat     = (float*) take((size_t)BLn * D * 4);
  bf16_t* segm_bf  = (bf16_t*)take((size_t)BLn * D * 2);
  int*    hard_i   = (int*)   take((size_t)BLn * 4);
  int*    segstart = (int*)   take((size_t)B * (L + 1) * 4);
  int*    nseg     = (int*)   take((size_t)B * 4);
  (void)ws_size; (void)n_in; (void)out_size;

  // 1) norms + bf16 conversions
  rownorm_cvt_kernel<<<BLn, 256, 0, stream>>>(hidden, hid_bf, nh_bf, rnorm, D);
  // 2) folded boundary weight M^T and bf16 weights
  mt_prep_kernel<<<(D * D + 255) / 256, 256, 0, stream>>>(Wqb, Wkb, Mt_bf, D);
  cvt_bf16_kernel<<<512, 256, 0, stream>>>(Wq, Wq_bf, D * D);
  cvt_bf16_kernel<<<512, 256, 0, stream>>>(Wk, Wk_bf, D * D);
  cvt_bf16_kernel<<<512, 256, 0, stream>>>(Wv, Wv_bf, D * D);
  // 3) WMMA GEMMs: y = nh @ M ; K = h @ Wk^T ; V = h @ Wv^T
  dim3 ggrid(D / 64, BLn / 128);
  gemm_bf16_wmma_kernel<<<ggrid, 256, 0, stream>>>(nh_bf, Mt_bf, yb, BLn);
  gemm_bf16_wmma_kernel<<<ggrid, 256, 0, stream>>>(hid_bf, Wk_bf, Kmat, BLn);
  gemm_bf16_wmma_kernel<<<ggrid, 256, 0, stream>>>(hid_bf, Wv_bf, Vmat, BLn);
  // 4) hard boundary bits
  boundary_hard_kernel<<<BLn / 8, 256, 0, stream>>>(yb, hidden, rnorm, noise_u,
                                                    hard_i, D);
  // 5) segment scan
  seg_scan_kernel<<<B, 1024, 0, stream>>>(hard_i, segstart, nseg);
  // 6) segment means
  seg_mean_kernel<<<dim3(L, B), 128, 0, stream>>>(hidden, segstart, nseg, segm_bf, D);
  // 7) Q = segmean @ Wq^T (WMMA)
  gemm_bf16_wmma_kernel<<<ggrid, 256, 0, stream>>>(segm_bf, Wq_bf, Qmat, BLn);
  // 8) segment-local attention pooling -> pooled output
  attn_pool_kernel<<<dim3(L, B), 256, 0, stream>>>(Qmat, Kmat, Vmat, segstart,
                                                   nseg, out, D);
  // 9) loss + counters appended after pooled [B, L, D]
  loss_kernel<<<1, 32, 0, stream>>>(nseg, out + (size_t)BLn * D);
}
